// BipartiteGNNEncoder_7026566496898
// MI455X (gfx1250) — matrix operand order repaired and checked by reading, not compiled, not run on hardware
//
#include <hip/hip_runtime.h>
#include <hip/hip_bf16.h>

// ---------------------------------------------------------------------------
// Bipartite GraphSAGE (2 layers) for MI455X / gfx1250.
//
// Roofline: edge aggregation dominates (3.2M edges x 64 feats x f32 gather +
// atomic scatter-add; feature tables are <=25.6MB so gathers/atomics resolve
// in the 192MB L2). Dense GEMMs (~3.4 GFLOP total) use
// v_wmma_f32_16x16x32_f16: f16 operands, f32 accumulation, fused
// (mean@Wl + x_dst@Wr + bias, ReLU). All GEMM shape/flag parameters are
// template constants so the WMMA chains and epilogue fully unroll with
// immediate-offset stores.
// ---------------------------------------------------------------------------

typedef __attribute__((ext_vector_type(16))) _Float16 v16h;
typedef __attribute__((ext_vector_type(8)))  _Float16 v8h;
typedef __attribute__((ext_vector_type(8)))  float    v8f;

#define NS_   100000
#define NV_   20000
#define EDGES 3200000

// ----- A-operand load: 16-bit A 16x32 lane layout (ISA 7.12.2) --------------
// lanes 0-15 : row M=lane,    halfs = K[0..7]  ++ K[16..23]
// lanes 16-31: row M=lane-16, halfs = K[8..15] ++ K[24..31]
// Caller pre-offsets the row pointer by half_sel*8; we load [p..p+7] and
// [p+16..p+23].
__device__ __forceinline__ v16h load_a16(const _Float16* p) {
  v8h lo = *(const v8h*)(p);
  v8h hi = *(const v8h*)(p + 16);
  v16h r;
#pragma unroll
  for (int i = 0; i < 8; ++i) { r[i] = lo[i]; r[i + 8] = hi[i]; }
  return r;
}

// ----- Pack weight f32 [K x N] into per-lane B operand image ----------------
// Packed element ((kb*ntiles + nt)*32 + lane)*16 + t  =  f16( W[kb*32 + (t&7) +
// 8*(lane>>4) + 16*(t>>3)][nt*16 + (lane&15)] ), zero padded past K. Matches the dense
// 16-bit B 32x16 lane layout (K striped 2-per-VGPR down each column).
__global__ void pack_w(const float* __restrict__ W, _Float16* __restrict__ dst,
                       int K, int N) {
  int kb_count = (K + 31) >> 5;
  int ntiles   = N >> 4;
  int total    = kb_count * ntiles * 512;
  int idx = blockIdx.x * blockDim.x + threadIdx.x;
  if (idx >= total) return;
  int t    = idx & 15;
  int lane = (idx >> 4) & 31;
  int tile = idx >> 9;
  int nt   = tile % ntiles;
  int kb   = tile / ntiles;
  int k = (kb << 5) + ((lane >> 4) << 3) + (t & 7) + ((t >> 3) << 4);
  int n = (nt << 4) + (lane & 15);
  float v = (k < K) ? W[k * N + n] : 0.f;
  dst[idx] = (_Float16)v;
}

// ----- Pad raw input rows (Kin=10/9) to f16 [M x 32] ------------------------
__global__ void pad_rows_h(const float* __restrict__ X, _Float16* __restrict__ Y,
                           int M, int Kin) {
  int idx = blockIdx.x * blockDim.x + threadIdx.x;
  if (idx >= M * 32) return;
  int c = idx & 31;
  int r = idx >> 5;
  Y[idx] = (_Float16)((c < Kin) ? X[r * Kin + c] : 0.f);
}

__global__ void zero_f32(float* __restrict__ p, int n) {
  int i = blockIdx.x * blockDim.x + threadIdx.x;
  if (i < n) p[i] = 0.f;
}

// ----- Edge scatter: agg[dst] += x[src], cnt[dst] += 1 ----------------------
// One thread per (edge, 4-feature group): float4 gather (hits L2) + 4
// hardware f32 atomics (global_atomic_add_f32, resolved at L2).
__global__ void edge_scatter(const float* __restrict__ xsrc,
                             const int* __restrict__ sI,
                             const int* __restrict__ dI,
                             float* __restrict__ agg, float* __restrict__ cnt,
                             int E) {
  int tid = blockIdx.x * blockDim.x + threadIdx.x;
  if (tid >= E * 16) return;
  int e  = tid >> 4;
  int fg = tid & 15;
  int s = sI[e];
  int d = dI[e];
  const float4 v = *(const float4*)(xsrc + ((size_t)s << 6) + (fg << 2));
  float* ap = agg + ((size_t)d << 6) + (fg << 2);
  unsafeAtomicAdd(ap + 0, v.x);
  unsafeAtomicAdd(ap + 1, v.y);
  unsafeAtomicAdd(ap + 2, v.z);
  unsafeAtomicAdd(ap + 3, v.w);
  if (fg == 0) unsafeAtomicAdd(cnt + d, 1.0f);
}

// ----- mean = agg / max(cnt,1), emitted as f16 GEMM A operand ---------------
__global__ void finalize_mean(const float* __restrict__ agg,
                              const float* __restrict__ cnt,
                              _Float16* __restrict__ mh, int total) {
  int i = blockIdx.x * blockDim.x + threadIdx.x;
  if (i >= total) return;
  float c = cnt[i >> 6];
  mh[i] = (_Float16)(agg[i] / fmaxf(c, 1.f));
}

// ----- Fused SAGE GEMM: out = act( A1@B1 + A2@B2 + bias ) -------------------
// One wave per 16-row tile. NT (N/16), K-block counts, ReLU and f16-copy
// flags are compile-time: the nt loop and wmma chains fully unroll, the
// epilogue is branch-free and store offsets fold to immediates.
// M % 16 == 0 always, so EXEC is all ones for every WMMA (ISA requirement).
template <int K1B, int K2B, int NT, bool RELU, bool WRITE_H>
__global__ void sage_gemm(const _Float16* __restrict__ A1, int lda1,
                          const _Float16* __restrict__ A2, int lda2,
                          const _Float16* __restrict__ B1,
                          const _Float16* __restrict__ B2,
                          const float* __restrict__ bias,
                          float* __restrict__ outF, _Float16* __restrict__ outH,
                          int M) {
  constexpr int N = NT * 16;
  int wave  = blockIdx.x * (blockDim.x >> 5) + (threadIdx.x >> 5);
  int tiles = M >> 4;
  if (wave >= tiles) return;                 // wave-uniform: EXEC stays full
  int lane     = threadIdx.x & 31;
  int half_sel = lane >> 4;
  int col_l    = lane & 15;
  int row      = (wave << 4) + col_l;
  const _Float16* a1p = A1 + (size_t)row * lda1 + (half_sel << 3);
  const _Float16* a2p = (K2B > 0) ? (A2 + (size_t)row * lda2 + (half_sel << 3))
                                  : nullptr;
  int r0 = (wave << 4) + (half_sel << 3);

#pragma unroll
  for (int nt = 0; nt < NT; ++nt) {
    v8f c = {};
    const _Float16* b1t = B1 + (nt << 9) + (lane << 4);
#pragma unroll
    for (int kb = 0; kb < K1B; ++kb) {
      v16h a = load_a16(a1p + (kb << 5));
      v16h b = *(const v16h*)(b1t + kb * (NT << 9));
      c = __builtin_amdgcn_wmma_f32_16x16x32_f16(false, a, false, b,
                                                 (short)0, c, false, false);
    }
    if (K2B > 0) {
      const _Float16* b2t = B2 + (nt << 9) + (lane << 4);
#pragma unroll
      for (int kb = 0; kb < K2B; ++kb) {
        v16h a = load_a16(a2p + (kb << 5));
        v16h b = *(const v16h*)(b2t + kb * (NT << 9));
        c = __builtin_amdgcn_wmma_f32_16x16x32_f16(false, a, false, b,
                                                   (short)0, c, false, false);
      }
    }
    int col  = (nt << 4) + col_l;
    float bv = bias[col];
    // C layout: VGPR v -> (M = r0 + v, N = col). One base pointer per tile;
    // v*N folds into the store-immediate offset.
    float*    op = outF + (size_t)r0 * N + col;
    _Float16* oh = WRITE_H ? (outH + (size_t)r0 * N + col) : nullptr;
#pragma unroll
    for (int v = 0; v < 8; ++v) {
      float val = c[v] + bv;
      if (RELU) val = fmaxf(val, 0.f);
      op[v * N] = val;
      if (WRITE_H) oh[v * N] = (_Float16)val;
    }
  }
}

template <int K1B, int K2B, int NT, bool RELU, bool WRITE_H>
static void run_gemm(hipStream_t s, const _Float16* A1, int lda1,
                     const _Float16* A2, int lda2, const _Float16* B1,
                     const _Float16* B2, const float* bias, float* outF,
                     _Float16* outH, int M) {
  int tiles  = M >> 4;
  int blocks = (tiles + 7) / 8;              // 8 wave32s / block
  sage_gemm<K1B, K2B, NT, RELU, WRITE_H>
      <<<blocks, 256, 0, s>>>(A1, lda1, A2, lda2, B1, B2, bias, outF, outH, M);
}

static inline int cdiv(long long a, long long b) { return (int)((a + b - 1) / b); }

extern "C" void kernel_launch(void* const* d_in, const int* in_sizes, int n_in,
                              void* d_out, int out_size, void* d_ws,
                              size_t ws_size, hipStream_t stream) {
  const float* x_site      = (const float*)d_in[0];
  const float* x_vendor    = (const float*)d_in[1];
  const int*   src         = (const int*)d_in[2];
  const int*   dst         = (const int*)d_in[3];
  const float* W_site_in   = (const float*)d_in[4];
  const float* b_site_in   = (const float*)d_in[5];
  const float* W_vendor_in = (const float*)d_in[6];
  const float* b_vendor_in = (const float*)d_in[7];
  const float* Wl1sv = (const float*)d_in[8];
  const float* bl1sv = (const float*)d_in[9];
  const float* Wr1sv = (const float*)d_in[10];
  const float* Wl1vs = (const float*)d_in[11];
  const float* bl1vs = (const float*)d_in[12];
  const float* Wr1vs = (const float*)d_in[13];
  const float* Wl2sv = (const float*)d_in[14];
  const float* bl2sv = (const float*)d_in[15];
  const float* Wr2sv = (const float*)d_in[16];
  const float* Wl2vs = (const float*)d_in[17];
  const float* bl2vs = (const float*)d_in[18];
  const float* Wr2vs = (const float*)d_in[19];

  const int NS = NS_, NV = NV_, E = EDGES;

  char*  ws  = (char*)d_ws;
  size_t off = 0;
  auto alloc = [&](size_t bytes) -> void* {
    void* p = (void*)(ws + off);
    off += (bytes + 255) & ~(size_t)255;
    return p;
  };

  // f32 activations / aggregation scratch (~142 MB total workspace)
  float* xs_f32  = (float*)alloc((size_t)NS * 64 * 4);
  float* xv_f32  = (float*)alloc((size_t)NV * 64 * 4);
  float* xs1_f32 = (float*)alloc((size_t)NS * 64 * 4);
  float* xv1_f32 = (float*)alloc((size_t)NV * 64 * 4);
  float* agg_s   = (float*)alloc((size_t)NS * 64 * 4);
  float* agg_v   = (float*)alloc((size_t)NV * 64 * 4);
  float* cnt_s   = (float*)alloc((size_t)NS * 4);
  float* cnt_v   = (float*)alloc((size_t)NV * 4);
  // f16 operand buffers
  _Float16* xsite_h   = (_Float16*)alloc((size_t)NS * 32 * 2);
  _Float16* xvendor_h = (_Float16*)alloc((size_t)NV * 32 * 2);
  _Float16* xs_h      = (_Float16*)alloc((size_t)NS * 64 * 2);
  _Float16* xv_h      = (_Float16*)alloc((size_t)NV * 64 * 2);
  _Float16* xs1_h     = (_Float16*)alloc((size_t)NS * 64 * 2);
  _Float16* xv1_h     = (_Float16*)alloc((size_t)NV * 64 * 2);
  _Float16* mean_s_h  = (_Float16*)alloc((size_t)NS * 64 * 2);
  _Float16* mean_v_h  = (_Float16*)alloc((size_t)NV * 64 * 2);
  // packed weights (B operand lane images)
  _Float16* pWsi   = (_Float16*)alloc(2048 * 2);  // K=10->32, N=64
  _Float16* pWvi   = (_Float16*)alloc(2048 * 2);  // K=9 ->32, N=64
  _Float16* pWl1sv = (_Float16*)alloc(4096 * 2);
  _Float16* pWr1sv = (_Float16*)alloc(4096 * 2);
  _Float16* pWl1vs = (_Float16*)alloc(4096 * 2);
  _Float16* pWr1vs = (_Float16*)alloc(4096 * 2);
  _Float16* pWl2sv = (_Float16*)alloc(2048 * 2);
  _Float16* pWr2sv = (_Float16*)alloc(2048 * 2);
  _Float16* pWl2vs = (_Float16*)alloc(2048 * 2);
  _Float16* pWr2vs = (_Float16*)alloc(2048 * 2);

  auto pw = [&](const float* W, int K, int N, _Float16* dstp) {
    int total = ((K + 31) >> 5) * (N >> 4) * 512;
    pack_w<<<cdiv(total, 256), 256, 0, stream>>>(W, dstp, K, N);
  };
  auto zz = [&](float* p, int n) {
    zero_f32<<<cdiv(n, 256), 256, 0, stream>>>(p, n);
  };

  // ---- weight packing --------------------------------------------------
  pw(W_site_in, 10, 64, pWsi);    pw(W_vendor_in, 9, 64, pWvi);
  pw(Wl1sv, 64, 64, pWl1sv);      pw(Wr1sv, 64, 64, pWr1sv);
  pw(Wl1vs, 64, 64, pWl1vs);      pw(Wr1vs, 64, 64, pWr1vs);
  pw(Wl2sv, 64, 32, pWl2sv);      pw(Wr2sv, 64, 32, pWr2sv);
  pw(Wl2vs, 64, 32, pWl2vs);      pw(Wr2vs, 64, 32, pWr2vs);

  // ---- input projection: xs = x_site@Wsi + b ; xv likewise -------------
  pad_rows_h<<<cdiv((size_t)NS * 32, 256), 256, 0, stream>>>(x_site, xsite_h, NS, 10);
  pad_rows_h<<<cdiv((size_t)NV * 32, 256), 256, 0, stream>>>(x_vendor, xvendor_h, NV, 9);
  run_gemm<1, 0, 4, false, true>(stream, xsite_h, 32, nullptr, 0, pWsi, nullptr,
                                 b_site_in, xs_f32, xs_h, NS);
  run_gemm<1, 0, 4, false, true>(stream, xvendor_h, 32, nullptr, 0, pWvi, nullptr,
                                 b_vendor_in, xv_f32, xv_h, NV);

  // ---- layer 1 aggregation ---------------------------------------------
  zz(agg_v, NV * 64); zz(cnt_v, NV);
  edge_scatter<<<cdiv((long long)E * 16, 256), 256, 0, stream>>>(
      xs_f32, src, dst, agg_v, cnt_v, E);
  finalize_mean<<<cdiv(NV * 64, 256), 256, 0, stream>>>(agg_v, cnt_v, mean_v_h, NV * 64);
  zz(agg_s, NS * 64); zz(cnt_s, NS);
  edge_scatter<<<cdiv((long long)E * 16, 256), 256, 0, stream>>>(
      xv_f32, dst, src, agg_s, cnt_s, E);
  finalize_mean<<<cdiv(NS * 64, 256), 256, 0, stream>>>(agg_s, cnt_s, mean_s_h, NS * 64);

  // ---- layer 1 fused SAGE GEMMs (relu) ---------------------------------
  run_gemm<2, 2, 4, true, true>(stream, mean_v_h, 64, xv_h, 64, pWl1sv, pWr1sv,
                                bl1sv, xv1_f32, xv1_h, NV);
  run_gemm<2, 2, 4, true, true>(stream, mean_s_h, 64, xs_h, 64, pWl1vs, pWr1vs,
                                bl1vs, xs1_f32, xs1_h, NS);

  // ---- layer 2 aggregation ---------------------------------------------
  zz(agg_v, NV * 64); zz(cnt_v, NV);
  edge_scatter<<<cdiv((long long)E * 16, 256), 256, 0, stream>>>(
      xs1_f32, src, dst, agg_v, cnt_v, E);
  finalize_mean<<<cdiv(NV * 64, 256), 256, 0, stream>>>(agg_v, cnt_v, mean_v_h, NV * 64);
  zz(agg_s, NS * 64); zz(cnt_s, NS);
  edge_scatter<<<cdiv((long long)E * 16, 256), 256, 0, stream>>>(
      xv1_f32, dst, src, agg_s, cnt_s, E);
  finalize_mean<<<cdiv(NS * 64, 256), 256, 0, stream>>>(agg_s, cnt_s, mean_s_h, NS * 64);

  // ---- layer 2 fused SAGE GEMMs -> d_out (concat [site; vendor]) -------
  float* outF = (float*)d_out;
  run_gemm<2, 2, 2, true, false>(stream, mean_s_h, 64, xs1_h, 64, pWl2vs, pWr2vs,
                                 bl2vs, outF, nullptr, NS);
  run_gemm<2, 2, 2, true, false>(stream, mean_v_h, 64, xv1_h, 64, pWl2sv, pWr2sv,
                                 bl2sv, outF + (size_t)NS * 32, nullptr, NV);
}
